// GATMemEfficient_5497558138997
// MI455X (gfx1250) — matrix operand order repaired and checked by reading, not compiled, not run on hardware
//
#include <hip/hip_runtime.h>
#include <hip/hip_fp16.h>
#include <math.h>

typedef __attribute__((ext_vector_type(16))) _Float16 v16h;
typedef __attribute__((ext_vector_type(8)))  float    v8f;

// ---------------------------------------------------------------- utilities

__global__ void fill_kernel(float* __restrict__ p, float v, long long n) {
    long long i = (long long)blockIdx.x * blockDim.x + threadIdx.x;
    if (i < n) p[i] = v;
}

__global__ void cvt_f16(const float* __restrict__ in, _Float16* __restrict__ o, long long n) {
    long long i = (long long)blockIdx.x * blockDim.x + threadIdx.x;
    if (i < n) o[i] = (_Float16)in[i];
}

// Wt[col*K + k] = col < Nc ? W[k*Nc + col] : 0   for col in [0,Nw), k in [0,K); f16 out
__global__ void transpose_pad_f16(const float* __restrict__ W, _Float16* __restrict__ Wt,
                                  int K, int Nc, int Nw) {
    int i = blockIdx.x * blockDim.x + threadIdx.x;
    if (i >= Nw * K) return;
    int col = i / K, k = i % K;
    Wt[i] = (col < Nc) ? (_Float16)W[(size_t)k * Nc + col] : (_Float16)0.f;
}

__device__ __forceinline__ void atomicMaxFloat(float* addr, float val) {
    // signed/unsigned split; buffer initialized to -inf
    if (val >= 0.f) atomicMax((int*)addr, __float_as_int(val));
    else            atomicMin((unsigned int*)addr, __float_as_uint(val));
}

__device__ __forceinline__ void edge_pair(const int* __restrict__ ei,
                                          long long eidx, long long E_,
                                          int& s, int& d) {
    if (eidx < E_) { s = ei[eidx]; d = ei[E_ + eidx]; }   // edge_index[0]=src, [1]=dst
    else           { s = d = (int)(eidx - E_); }           // self loop
}

// ------------------------------------------------------ WMMA GEMM (f16 in, f32 acc)
// C[M x Nc] = A[M x K] * W[K x Nc]; A stored f16 row-major, W pre-transposed+padded
// f16 as Wt[Nw x K].  M % 16 == 0, K % 32 == 0.
// block = 128 threads (4 waves); each wave computes NT adjacent 16x16 tiles of one
// 16-row band, reusing a single A fragment across the NT WMMAs per k-step.
template<int NT>
__global__ void __launch_bounds__(128)
gemm_wmma_f16(const _Float16* __restrict__ A,
              const _Float16* __restrict__ Wt,
              float* __restrict__ C,
              int M, int K, int Nc) {
    const int lane = threadIdx.x & 31;
    const int wv   = threadIdx.x >> 5;
    const int m0   = blockIdx.x * 16;
    const int n0   = wv * (16 * NT);
    const int hi   = lane >> 4;            // 0: lanes 0-15, 1: lanes 16-31
    const int l16  = lane & 15;

    const _Float16* Ar = A + (size_t)(m0 + l16) * K;
    v8f acc[NT];
    #pragma unroll
    for (int t = 0; t < NT; ++t) acc[t] = (v8f){};

    for (int k0 = 0; k0 < K; k0 += 32) {
        if (k0 + 32 < K) __builtin_prefetch(Ar + k0 + 32, 0, 3);   // WGP-scope global_prefetch_b8
        // A fragment: contiguous k = 16*hi + j (matches B); 16 contiguous
        // halfs per lane -> 2x global_load_b128, no conversion.
        v16h af;
        #pragma unroll
        for (int j = 0; j < 8; ++j) af[j]     = Ar[k0 + hi * 16 + j];
        #pragma unroll
        for (int j = 0; j < 8; ++j) af[8 + j] = Ar[k0 + hi * 16 + 8 + j];
        #pragma unroll
        for (int t = 0; t < NT; ++t) {
            // B fragment: 32x16 (KxN); 16 contiguous halfs per lane -> 2x b128.
            const _Float16* Wr = Wt + (size_t)(n0 + t * 16 + l16) * K + k0 + hi * 16;
            v16h bf;
            #pragma unroll
            for (int j = 0; j < 16; ++j) bf[j] = Wr[j];
            acc[t] = __builtin_amdgcn_wmma_f32_16x16x32_f16(false, af, false, bf,
                                                            (short)0, acc[t], false, false);
        }
    }
    #pragma unroll
    for (int t = 0; t < NT; ++t) {
        int bcol = n0 + t * 16 + l16;
        if (bcol < Nc) {
            #pragma unroll
            for (int r = 0; r < 8; ++r)          // C/D layout: VGPR r -> M=r (+8 hi lanes)
                C[(size_t)(m0 + r + hi * 8) * Nc + bcol] = acc[t][r];
        }
    }
}

// ------------------------------------------------------ attention coefficients
__global__ void att_coeff(const float* __restrict__ h,
                          const float* __restrict__ att_s,
                          const float* __restrict__ att_d,
                          float* __restrict__ as_, float* __restrict__ ad_,
                          long long N_, int H, int Cc) {
    long long i = (long long)blockIdx.x * blockDim.x + threadIdx.x;
    if (i >= N_ * H) return;
    int hh = (int)(i % H);
    const float* hp = h + i * Cc;          // [n,h,c] contiguous
    float s = 0.f, d = 0.f;
    for (int c = 0; c < Cc; ++c) {
        float v = hp[c];
        s += v * att_s[hh * Cc + c];
        d += v * att_d[hh * Cc + c];
    }
    as_[i] = s; ad_[i] = d;
}

// ------------------------------------------------------ edge passes
__global__ void edge_max(const int* __restrict__ ei, long long E_, long long NE,
                         const float* __restrict__ as_, const float* __restrict__ ad_,
                         float* __restrict__ mx, int H) {
    long long gid = (long long)blockIdx.x * blockDim.x + threadIdx.x;
    if (gid >= NE * H) return;
    long long eidx = gid / H; int hh = (int)(gid % H);
    int s, d; edge_pair(ei, eidx, E_, s, d);
    float e = as_[(size_t)s * H + hh] + ad_[(size_t)d * H + hh];
    e = (e > 0.f) ? e : 0.2f * e;                       // leaky relu
    atomicMaxFloat(&mx[(size_t)d * H + hh], e);
}

// exp(e - max) stored per (edge,head) + atomic segment sum
__global__ void edge_sum(const int* __restrict__ ei, long long E_, long long NE,
                         const float* __restrict__ as_, const float* __restrict__ ad_,
                         const float* __restrict__ mx, float* __restrict__ sm,
                         float* __restrict__ ea, int H) {
    long long gid = (long long)blockIdx.x * blockDim.x + threadIdx.x;
    if (gid >= NE * H) return;
    long long eidx = gid / H; int hh = (int)(gid % H);
    int s, d; edge_pair(ei, eidx, E_, s, d);
    float e = as_[(size_t)s * H + hh] + ad_[(size_t)d * H + hh];
    e = (e > 0.f) ? e : 0.2f * e;
    float v = expf(e - mx[(size_t)d * H + hh]);
    ea[gid] = v;
    atomicAdd(&sm[(size_t)d * H + hh], v);
}

// ea[e,h] /= (sum[dst,h] + eps)  -> final alpha per (edge,head)
__global__ void edge_norm(const int* __restrict__ ei, long long E_, long long NE,
                          const float* __restrict__ sm, float* __restrict__ ea, int H) {
    long long gid = (long long)blockIdx.x * blockDim.x + threadIdx.x;
    if (gid >= NE * H) return;
    long long eidx = gid / H; int hh = (int)(gid % H);
    int s, d; edge_pair(ei, eidx, E_, s, d);
    ea[gid] = ea[gid] / (sm[(size_t)d * H + hh] + 1e-16f);
}

// one thread per (edge, channel): minimal traffic -> alpha broadcast + coalesced
// h[src] gather + coalesced global_atomic_add_f32 scatter (L2-resident).
__global__ void edge_agg(const int* __restrict__ ei, long long E_, long long NE,
                         const float* __restrict__ ea,
                         const float* __restrict__ h, float* __restrict__ out,
                         int H, int Cc) {
    const int HC = H * Cc;
    long long gid = (long long)blockIdx.x * blockDim.x + threadIdx.x;
    if (gid >= NE * HC) return;
    long long eidx = gid / HC; int ch = (int)(gid % HC);
    int hh = ch / Cc;
    int s, d; edge_pair(ei, eidx, E_, s, d);
    float alpha = ea[eidx * H + hh];
    atomicAdd(&out[(size_t)d * HC + ch], h[(size_t)s * HC + ch] * alpha);
}

// ------------------------------------------------------ bias + batchnorm + elu -> f16
// Reads fp32 aggregation, writes f16 activations (consumed only by the next GEMM).
__global__ void bias_bn_elu_f16(const float* __restrict__ x, _Float16* __restrict__ o,
                                const float* __restrict__ bias,
                                const float* __restrict__ g, const float* __restrict__ b,
                                const float* __restrict__ m, const float* __restrict__ v,
                                long long total, int F) {
    long long i = (long long)blockIdx.x * blockDim.x + threadIdx.x;
    if (i >= total) return;
    int f = (int)(i % F);
    float val = x[i] + bias[f];
    val = (val - m[f]) * (g[f] * rsqrtf(v[f] + 1e-5f)) + b[f];
    val = (val > 0.f) ? val : (expf(val) - 1.f);        // elu
    o[i] = (_Float16)val;
}

// ------------------------------------------------------ bias + log_softmax over F
__global__ void bias_logsoftmax(const float* __restrict__ agg, const float* __restrict__ bias,
                                float* __restrict__ out, long long N_, int F) {
    long long n = (long long)blockIdx.x * blockDim.x + threadIdx.x;
    if (n >= N_) return;
    const float* row = agg + n * F;
    float mx = -INFINITY;
    for (int f = 0; f < F; ++f) mx = fmaxf(mx, row[f] + bias[f]);
    float s = 0.f;
    for (int f = 0; f < F; ++f) s += expf(row[f] + bias[f] - mx);
    float ls = logf(s);
    for (int f = 0; f < F; ++f) out[n * F + f] = row[f] + bias[f] - mx - ls;
}

// ---------------------------------------------------------------- driver

static inline unsigned cdiv(long long a, long long b) { return (unsigned)((a + b - 1) / b); }

extern "C" void kernel_launch(void* const* d_in, const int* in_sizes, int n_in,
                              void* d_out, int out_size, void* d_ws, size_t ws_size,
                              hipStream_t stream) {
    const float* x      = (const float*)d_in[0];
    const int*   ei     = (const int*)  d_in[1];   // [2,E]: row0=src, row1=dst (int32)
    const float* W1     = (const float*)d_in[2];
    const float* att_s1 = (const float*)d_in[3];
    const float* att_d1 = (const float*)d_in[4];
    const float* b1     = (const float*)d_in[5];
    const float* bn1g   = (const float*)d_in[6];
    const float* bn1b   = (const float*)d_in[7];
    const float* bn1m   = (const float*)d_in[8];
    const float* bn1v   = (const float*)d_in[9];
    const float* W2     = (const float*)d_in[10];
    const float* att_s2 = (const float*)d_in[11];
    const float* att_d2 = (const float*)d_in[12];
    const float* b2     = (const float*)d_in[13];
    const float* bn2g   = (const float*)d_in[14];
    const float* bn2b   = (const float*)d_in[15];
    const float* bn2m   = (const float*)d_in[16];
    const float* bn2v   = (const float*)d_in[17];
    const float* W3     = (const float*)d_in[18];
    const float* att_s3 = (const float*)d_in[19];
    const float* att_d3 = (const float*)d_in[20];
    const float* b3     = (const float*)d_in[21];
    float* out = (float*)d_out;

    const long long N_ = in_sizes[0] / 128;   // 100000
    const long long E_ = in_sizes[1] / 2;     // 1600000
    const long long NE = E_ + N_;             // edges incl. self loops

    // ---- workspace layout (float units)
    float* ws  = (float*)d_ws;
    float* h1  = ws;                     // N*256 fp32 (reused: h2/h3, x2h, agg2/3)
    float* xb  = ws + N_ * 256;          // N*256 fp32 (Xh, agg1, x3h)
    float* sm0 = ws + 2 * N_ * 256;
    float* as1 = sm0;            float* ad1 = as1 + 4 * N_;
    float* m1  = ad1 + 4 * N_;   float* s1  = m1  + 4 * N_;
    float* as2 = s1  + 4 * N_;   float* ad2 = as2 + N_;
    float* m2  = ad2 + N_;       float* s2  = m2  + N_;
    float* as3 = s2  + N_;       float* ad3 = as3 + N_;
    float* m3  = ad3 + N_;       float* s3  = m3  + N_;
    _Float16* Wt1 = (_Float16*)(s3 + N_);   // 256*128 halfs (16384 floats)
    _Float16* Wt2 = Wt1 + 256 * 128;        // 64*256 halfs  (8192 floats)
    _Float16* Wt3 = Wt2 + 64 * 256;         // 64*64 halfs   (2048 floats; padded from 40)
    float* ealpha = s3 + N_ + 16384 + 8192 + 2048;   // NE*4 fp32 (per-edge alpha, worst case H=4)

    // f16 activation views carved from dead fp32 regions (stream-order safe):
    _Float16* Xh  = (_Float16*)xb;              // N*128 halfs; xb zero-filled only after gemm1
    _Float16* x2h = (_Float16*)(h1 + N_ * 128); // N*256 halfs; h1 upper half dead after edge_agg1
    _Float16* x3h = (_Float16*)xb;              // N*64 halfs; xb(agg1) dead after bn_elu1

    float* h2   = h1;                    // N*64
    float* agg2 = h1 + N_ * 64;          // N*64
    float* h3   = h1;                    // N*40
    float* agg3 = h1 + N_ * 64;          // N*40

    const int TB = 256;
    #define FILL(p, v, n) fill_kernel<<<cdiv((n), TB), TB, 0, stream>>>((p), (v), (n))

    // ---- one-time prep: f16 weights (transposed+padded) and f16 input
    transpose_pad_f16<<<cdiv(256 * 128, TB), TB, 0, stream>>>(W1, Wt1, 128, 256, 256);
    transpose_pad_f16<<<cdiv(64 * 256, TB), TB, 0, stream>>>(W2, Wt2, 256, 64, 64);
    transpose_pad_f16<<<cdiv(64 * 64, TB), TB, 0, stream>>>(W3, Wt3, 64, 40, 64);
    cvt_f16<<<cdiv(N_ * 128, TB), TB, 0, stream>>>(x, Xh, N_ * 128);

    // ============================= layer 1: H=4, C=64, in=128 =============================
    gemm_wmma_f16<4><<<dim3((unsigned)(N_ / 16), 1), 128, 0, stream>>>(Xh, Wt1, h1, (int)N_, 128, 256);
    att_coeff<<<cdiv(N_ * 4, TB), TB, 0, stream>>>(h1, att_s1, att_d1, as1, ad1, N_, 4, 64);
    FILL(m1, -INFINITY, 4 * N_);
    FILL(s1, 0.f, 4 * N_);
    FILL(xb, 0.f, N_ * 256);
    edge_max<<<cdiv(NE * 4, TB), TB, 0, stream>>>(ei, E_, NE, as1, ad1, m1, 4);
    edge_sum<<<cdiv(NE * 4, TB), TB, 0, stream>>>(ei, E_, NE, as1, ad1, m1, s1, ealpha, 4);
    edge_norm<<<cdiv(NE * 4, TB), TB, 0, stream>>>(ei, E_, NE, s1, ealpha, 4);
    edge_agg<<<cdiv(NE * 256, TB), TB, 0, stream>>>(ei, E_, NE, ealpha, h1, xb, 4, 64);
    bias_bn_elu_f16<<<cdiv(N_ * 256, TB), TB, 0, stream>>>(xb, x2h, b1, bn1g, bn1b, bn1m, bn1v,
                                                           N_ * 256, 256);

    // ============================= layer 2: H=1, C=64, in=256 =============================
    gemm_wmma_f16<1><<<dim3((unsigned)(N_ / 16), 1), 128, 0, stream>>>(x2h, Wt2, h2, (int)N_, 256, 64);
    att_coeff<<<cdiv(N_, TB), TB, 0, stream>>>(h2, att_s2, att_d2, as2, ad2, N_, 1, 64);
    FILL(m2, -INFINITY, N_);
    FILL(s2, 0.f, N_);
    FILL(agg2, 0.f, N_ * 64);
    edge_max<<<cdiv(NE, TB), TB, 0, stream>>>(ei, E_, NE, as2, ad2, m2, 1);
    edge_sum<<<cdiv(NE, TB), TB, 0, stream>>>(ei, E_, NE, as2, ad2, m2, s2, ealpha, 1);
    edge_norm<<<cdiv(NE, TB), TB, 0, stream>>>(ei, E_, NE, s2, ealpha, 1);
    edge_agg<<<cdiv(NE * 64, TB), TB, 0, stream>>>(ei, E_, NE, ealpha, h2, agg2, 1, 64);
    bias_bn_elu_f16<<<cdiv(N_ * 64, TB), TB, 0, stream>>>(agg2, x3h, b2, bn2g, bn2b, bn2m, bn2v,
                                                          N_ * 64, 64);

    // ============================= layer 3: H=1, C=40, in=64 ==============================
    gemm_wmma_f16<1><<<dim3((unsigned)(N_ / 16), 1), 128, 0, stream>>>(x3h, Wt3, h3, (int)N_, 64, 40);
    att_coeff<<<cdiv(N_, TB), TB, 0, stream>>>(h3, att_s3, att_d3, as3, ad3, N_, 1, 40);
    FILL(m3, -INFINITY, N_);
    FILL(s3, 0.f, N_);
    FILL(agg3, 0.f, N_ * 40);
    edge_max<<<cdiv(NE, TB), TB, 0, stream>>>(ei, E_, NE, as3, ad3, m3, 1);
    edge_sum<<<cdiv(NE, TB), TB, 0, stream>>>(ei, E_, NE, as3, ad3, m3, s3, ealpha, 1);
    edge_norm<<<cdiv(NE, TB), TB, 0, stream>>>(ei, E_, NE, s3, ealpha, 1);
    edge_agg<<<cdiv(NE * 40, TB), TB, 0, stream>>>(ei, E_, NE, ealpha, h3, agg3, 1, 40);
    bias_logsoftmax<<<cdiv(N_, TB), TB, 0, stream>>>(agg3, b3, out, N_, 40);

    #undef FILL
}
